// StochasticGAT_9723805958349
// MI455X (gfx1250) — matrix operand order repaired and checked by reading, not compiled, not run on hardware
//
#include <hip/hip_runtime.h>
#include <hip/hip_bf16.h>

// ---------------- problem constants (from reference) ----------------
#define NN   50000
#define EE   800000
#define DIM  128          // IN == HEADS*H1 == HEADS*H2 == EMB == 128
#define HEADS 4
#define HD   32           // per-head dim

typedef __attribute__((ext_vector_type(16))) __bf16 v16bf;
typedef __attribute__((ext_vector_type(8)))  float  v8f;
typedef __attribute__((ext_vector_type(4)))  float  f32x4;
typedef __attribute__((ext_vector_type(4)))  unsigned u32x4;

// order-preserving float<->uint encoding so atomicMax(u32) == float max
__device__ __forceinline__ unsigned fenc(float f) {
    unsigned u = __builtin_bit_cast(unsigned, f);
    return u ^ ((u >> 31) ? 0xFFFFFFFFu : 0x80000000u);
}
__device__ __forceinline__ float fdec(unsigned u) {
    u ^= ((u >> 31) ? 0x80000000u : 0xFFFFFFFFu);
    return __builtin_bit_cast(float, u);
}

// ---------------- weight prepack: fp32 [128x128] -> bf16 WMMA-B fragment order ----------------
// Packed index: ((kstep*8 + ntile)*32 + lane)*16 + j
//   maps element  k = kstep*32 + (j>>3)*16 + (lane>>4)*8 + (j&7) ,  col = ntile*16 + (lane&15)
// so each lane's v16bf B fragment is one contiguous 32-byte load (2 x b128).
__global__ __launch_bounds__(256)
void pack_b_kernel(const float* __restrict__ W, unsigned short* __restrict__ P)
{
    int t = blockIdx.x * 256 + threadIdx.x;     // 0 .. 16383
    if (t >= DIM * DIM) return;
    const int j    = t & 15;
    const int lane = (t >> 4) & 31;
    const int nt   = (t >> 9) & 7;
    const int ks   = t >> 12;                   // 0..3
    const int k    = ks * 32 + (j >> 3) * 16 + (lane >> 4) * 8 + (j & 7);
    const int col  = nt * 16 + (lane & 15);
    __bf16 b = (__bf16)W[k * DIM + col];        // native RNE cvt
    P[t] = __builtin_bit_cast(unsigned short, b);
}

// ---------------- WMMA GEMM: C[M x 128] = A[M x 128] * Bpacked + bias ----------------
__global__ __launch_bounds__(256)
void gemm128_bf16_wmma(const float* __restrict__ A,
                       const v16bf* __restrict__ Bp,
                       const float* __restrict__ bias,
                       float* __restrict__ C, int M)
{
    const int wave = threadIdx.x >> 5;
    const int lane = threadIdx.x & 31;
    const int l16  = lane & 15;
    const int hi   = lane >> 4;                 // half-wave select
    const int rowBase = blockIdx.x * 128 + wave * 16;

    v8f acc[8];
    #pragma unroll
    for (int i = 0; i < 8; ++i)
        #pragma unroll
        for (int j = 0; j < 8; ++j) acc[i][j] = 0.0f;

    int r = rowBase + l16;
    if (r >= M) r = M - 1;                      // clamp; stores are guarded
    const float* __restrict__ arow = A + (long)r * DIM;

    #pragma unroll
    for (int ks = 0; ks < 4; ++ks) {
        // --- A fragment: 16-bit A 16x32 layout; 4 x b128 loads + native cvt ---
        const int kbA = ks * 32 + hi * 8;
        const f32x4 f0 = *(const f32x4*)(arow + kbA);
        const f32x4 f1 = *(const f32x4*)(arow + kbA + 4);
        const f32x4 f2 = *(const f32x4*)(arow + kbA + 16);
        const f32x4 f3 = *(const f32x4*)(arow + kbA + 20);
        v16bf a;
        #pragma unroll
        for (int j = 0; j < 4; ++j) {
            a[j]      = (__bf16)f0[j];
            a[4 + j]  = (__bf16)f1[j];
            a[8 + j]  = (__bf16)f2[j];
            a[12 + j] = (__bf16)f3[j];
        }
        #pragma unroll
        for (int nt = 0; nt < 8; ++nt) {
            // --- B fragment: one contiguous 32B load in packed order ---
            const v16bf b = Bp[((ks * 8 + nt) << 5) + lane];
            acc[nt] = __builtin_amdgcn_wmma_f32_16x16x32_bf16(
                false, a, false, b, (short)0, acc[nt], false, false);
        }
    }

    #pragma unroll
    for (int nt = 0; nt < 8; ++nt) {
        const int col = nt * 16 + l16;
        const float badd = bias ? bias[col] : 0.0f;
        #pragma unroll
        for (int j = 0; j < 8; ++j) {
            const int row = rowBase + j + hi * 8;   // C/D layout: vgpr j -> M=j (+8 hi half)
            if (row < M) C[(long)row * DIM + col] = acc[nt][j] + badd;
        }
    }
}

// ---------------- per-node attention scores el/er (wave32 reduction, 1 wave = 1 head) ----------------
__global__ __launch_bounds__(128)
void attn_scores_kernel(const float* __restrict__ feat,
                        const float* __restrict__ al, const float* __restrict__ ar,
                        float* __restrict__ el, float* __restrict__ er)
{
    const int node = blockIdx.x;
    const int d = threadIdx.x;                  // 0..127, head = d>>5
    const float f = feat[(long)node * DIM + d];
    float a = f * al[d];
    float b = f * ar[d];
    #pragma unroll
    for (int m = 16; m >= 1; m >>= 1) {
        a += __shfl_xor(a, m, 32);
        b += __shfl_xor(b, m, 32);
    }
    if ((d & 31) == 0) {
        const int h = d >> 5;
        el[node * HEADS + h] = a;
        er[node * HEADS + h] = b;
    }
}

// ---------------- init m/z/acc ----------------
__global__ __launch_bounds__(256)
void init_kernel(float* __restrict__ acc, float* __restrict__ z, unsigned* __restrict__ m,
                 int n128, int n4)
{
    int t = blockIdx.x * 256 + threadIdx.x;
    if (t < n128) acc[t] = 0.0f;
    if (t < n4) { z[t] = 0.0f; m[t] = 0u; }     // 0u < fenc(x) for any finite x > -inf
}

// ---------------- edge pass 1: leaky_relu(el[src]+er[dst]) + segment max (thread/edge, x4 wide) ----
__global__ __launch_bounds__(256)
void edge_scores_kernel(const int* __restrict__ src, const int* __restrict__ dst,
                        const float* __restrict__ el, const float* __restrict__ er,
                        float* __restrict__ ebuf, unsigned* __restrict__ m)
{
    int e = blockIdx.x * 256 + threadIdx.x;
    if (e >= EE) return;
    const int s = src[e], d = dst[e];
    const f32x4 l = *(const f32x4*)(el + s * HEADS);
    const f32x4 r = *(const f32x4*)(er + d * HEADS);
    f32x4 v = l + r;
    #pragma unroll
    for (int h = 0; h < HEADS; ++h) v[h] = (v[h] > 0.0f) ? v[h] : v[h] * 0.2f;
    *(f32x4*)(ebuf + (long)e * HEADS) = v;
    #pragma unroll
    for (int h = 0; h < HEADS; ++h) atomicMax(&m[d * HEADS + h], fenc(v[h]));
}

// ---------------- edge pass 2: exp(e - m[dst]) + segment sum (thread/edge, x4 wide) ----------------
__global__ __launch_bounds__(256)
void edge_exp_kernel(const int* __restrict__ dst,
                     float* __restrict__ ebuf, const unsigned* __restrict__ m,
                     float* __restrict__ z)
{
    int e = blockIdx.x * 256 + threadIdx.x;
    if (e >= EE) return;
    const int d = dst[e];
    const u32x4 mu = *(const u32x4*)(m + d * HEADS);
    f32x4 v = *(const f32x4*)(ebuf + (long)e * HEADS);
    #pragma unroll
    for (int h = 0; h < HEADS; ++h) v[h] = __expf(v[h] - fdec(mu[h]));
    *(f32x4*)(ebuf + (long)e * HEADS) = v;
    #pragma unroll
    for (int h = 0; h < HEADS; ++h) atomicAdd(&z[d * HEADS + h], v[h]);
}

// ---------------- edge pass 3: acc[dst] += ex * feat[src] (unnormalized) ----------------
__global__ __launch_bounds__(256)
void edge_agg_kernel(const int* __restrict__ src, const int* __restrict__ dst,
                     const float* __restrict__ feat, const float* __restrict__ ebuf,
                     float* __restrict__ acc, long long total)
{
    long long t = (long long)blockIdx.x * 256 + threadIdx.x;
    if (t >= total) return;
    const int e = (int)(t >> 7);
    const int d = (int)(t & 127);
    const int h = d >> 5;
    const int s  = src[e];
    const int dd = dst[e];
    const float ex = ebuf[e * HEADS + h];
    atomicAdd(&acc[(long)dd * DIM + d], ex * feat[(long)s * DIM + d]);
}

// ---------------- finalize: out = acc / z + bias ----------------
__global__ __launch_bounds__(256)
void finalize_kernel(const float* __restrict__ acc, const float* __restrict__ z,
                     const float* __restrict__ bias, float* __restrict__ out, int n128)
{
    int t = blockIdx.x * 256 + threadIdx.x;
    if (t >= n128) return;
    const int d = t & 127;
    const int n = t >> 7;
    const float zv = z[n * HEADS + (d >> 5)];
    const float v = (zv > 0.0f) ? acc[t] / zv : 0.0f;   // empty dst segment -> 0
    out[t] = v + bias[d];
}

// ---------------- orchestration ----------------
extern "C" void kernel_launch(void* const* d_in, const int* in_sizes, int n_in,
                              void* d_out, int out_size, void* d_ws, size_t ws_size,
                              hipStream_t stream)
{
    const float* x   = (const float*)d_in[0];
    const int*   src = (const int*)  d_in[1];
    const int*   dst = (const int*)  d_in[2];
    const float* W1  = (const float*)d_in[3];
    const float* al1 = (const float*)d_in[4];
    const float* ar1 = (const float*)d_in[5];
    const float* b1  = (const float*)d_in[6];
    const float* W2  = (const float*)d_in[7];
    const float* al2 = (const float*)d_in[8];
    const float* ar2 = (const float*)d_in[9];
    const float* b2  = (const float*)d_in[10];
    const float* Wp  = (const float*)d_in[11];
    const float* bp  = (const float*)d_in[12];

    // workspace carve-up (all offsets 32B-aligned)
    float* feat = (float*)d_ws;                 // N*128
    float* acc  = feat + (long)NN * DIM;        // N*128
    float* h1   = acc  + (long)NN * DIM;        // N*128
    float* ebuf = h1   + (long)NN * DIM;        // E*4
    float* el   = ebuf + (long)EE * HEADS;      // N*4
    float* er   = el   + (long)NN * HEADS;      // N*4
    float* z    = er   + (long)NN * HEADS;      // N*4
    unsigned* m = (unsigned*)(z + (long)NN * HEADS);               // N*4
    unsigned short* w1b = (unsigned short*)(m + (long)NN * HEADS); // 128*128
    unsigned short* w2b = w1b + DIM * DIM;
    unsigned short* wpb = w2b + DIM * DIM;

    float* h2  = (float*)d_out;                 // output half 1: N*128
    float* prj = h2 + (long)NN * DIM;           // output half 2: N*128

    const int gemmBlocks = (NN + 127) / 128;    // 391
    const int n128 = NN * DIM;                  // 6.4M
    const int n4   = NN * HEADS;                // 200K
    const long long eTot = (long long)EE * DIM; // 102.4M
    const int initBlocks = (n128 + 255) / 256;
    const int eBlocks    = (EE + 255) / 256;
    const int eAggBlocks = (int)((eTot + 255) / 256);
    const int finBlocks  = (n128 + 255) / 256;

    // prepack weights to WMMA-fragment-ordered bf16
    pack_b_kernel<<<64, 256, 0, stream>>>(W1, w1b);
    pack_b_kernel<<<64, 256, 0, stream>>>(W2, w2b);
    pack_b_kernel<<<64, 256, 0, stream>>>(Wp, wpb);

    // ---------- layer 1 ----------
    gemm128_bf16_wmma<<<gemmBlocks, 256, 0, stream>>>(x, (const v16bf*)w1b, nullptr, feat, NN);
    attn_scores_kernel<<<NN, 128, 0, stream>>>(feat, al1, ar1, el, er);
    init_kernel<<<initBlocks, 256, 0, stream>>>(acc, z, m, n128, n4);
    edge_scores_kernel<<<eBlocks, 256, 0, stream>>>(src, dst, el, er, ebuf, m);
    edge_exp_kernel<<<eBlocks, 256, 0, stream>>>(dst, ebuf, m, z);
    edge_agg_kernel<<<eAggBlocks, 256, 0, stream>>>(src, dst, feat, ebuf, acc, eTot);
    finalize_kernel<<<finBlocks, 256, 0, stream>>>(acc, z, b1, h1, n128);

    // ---------- layer 2 ----------
    gemm128_bf16_wmma<<<gemmBlocks, 256, 0, stream>>>(h1, (const v16bf*)w2b, nullptr, feat, NN);
    attn_scores_kernel<<<NN, 128, 0, stream>>>(feat, al2, ar2, el, er);
    init_kernel<<<initBlocks, 256, 0, stream>>>(acc, z, m, n128, n4);
    edge_scores_kernel<<<eBlocks, 256, 0, stream>>>(src, dst, el, er, ebuf, m);
    edge_exp_kernel<<<eBlocks, 256, 0, stream>>>(dst, ebuf, m, z);
    edge_agg_kernel<<<eAggBlocks, 256, 0, stream>>>(src, dst, feat, ebuf, acc, eTot);
    finalize_kernel<<<finBlocks, 256, 0, stream>>>(acc, z, b2, h2, n128);

    // ---------- final projection (h2 @ Wp + bp) ----------
    gemm128_bf16_wmma<<<gemmBlocks, 256, 0, stream>>>(h2, (const v16bf*)wpb, bp, prj, NN);
}
